// CrystalGraphConvNet_19645180412352
// MI455X (gfx1250) — matrix operand order repaired and checked by reading, not compiled, not run on hardware
//
#include <hip/hip_runtime.h>
#include <hip/hip_bf16.h>
#include <hip/hip_fp16.h>

// ---------------------------------------------------------------------------
// CrystalGraphConvNet forward for MI455X (gfx1250, wave32, WMMA).
// Heavy convs (node conv / edge conv / lin conv) run as tap-pair GEMMs on
// v_wmma_f32_16x16x32_f16: M=cout, N=16-position tile, K=32 (= 2 taps x 16 cin).
// B-fragment LDS addresses are precomputed per lane (shared by conv1/conv2).
// ---------------------------------------------------------------------------

typedef __attribute__((ext_vector_type(16))) _Float16 v16h;
typedef __attribute__((ext_vector_type(8)))  _Float16 v8h;
typedef __attribute__((ext_vector_type(8)))  float    v8f;
typedef __attribute__((ext_vector_type(4)))  float    v4f;

#define DEV static __device__ __forceinline__

#define NNODE 4096
#define NEDGE 16384
#define NB    64
#define HH    7
#define WW    18
#define HWP   126      // 7*18
#define PPAD  128      // padded position count (slots 126,127 are zero)

DEV float sp_f(float x)  { return x > 20.f ? x : log1pf(expf(x)); }
DEV float elu_f(float x) { return x > 0.f ? x : expm1f(x); }
DEV float sig_f(float x) { return 1.f / (1.f + expf(-x)); }

DEV v8f zero8() {
  v8f z;
#pragma unroll
  for (int i = 0; i < 8; ++i) z[i] = 0.f;
  return z;
}

DEV v16h cat16(v8h lo, v8h hi) {
  v16h r;
#pragma unroll
  for (int i = 0; i < 8; ++i) { r[i] = lo[i]; r[i + 8] = hi[i]; }
  return r;
}

DEV v8f wmma16(v16h a, v16h b, v8f c) {
  // D = A(16x32 f16) * B(32x16 f16) + C(16x16 f32)
  return __builtin_amdgcn_wmma_f32_16x16x32_f16(false, a, false, b, (short)0, c,
                                                false, false);
}

// A fragment (16x32 f16).  Weights in LDS laid out [tap][cout][cin] halves,
// tap slot 9 zeroed.  Lane layout: lane<16 -> K 0..15,
// lane>=16 -> K 16..31; row M = lane&15.
// K = (tap within pair)*16 + cin  => two contiguous 16B LDS loads per lane.
DEV v16h load_a_frag(const _Float16* wlds, int ks, int coutBase, int coutStride,
                     int lane) {
  int m    = lane & 15;
  int t0   = ks * 2 + (lane >> 4);
  if (t0 > 8) t0 = 9;  // zero slot
  const v8h* p0 = (const v8h*)(wlds + (size_t)(t0 * coutStride + coutBase + m) * 16);
  return cat16(p0[0], p0[1]);
}

// B fragment (32x16 f16) from a [pos(128)][cin(16)] half map at a precomputed
// byte offset (slot 126 provides zeros for out-of-range / pad taps).
DEV v16h load_b_at(const _Float16* flds, int byteOff) {
  const v8h* q = (const v8h*)((const char*)flds + byteOff);
  return cat16(q[0], q[1]);
}

// Precompute the per-lane shifted-position byte offset for (tile, kstep).
// Lane = B column (position in tile); lane<16 covers tap 2*ks, lane>=16 covers
// tap 2*ks+1.  Invalid (out-of-image, p>=126, or pad tap 9) -> zero slot 126.
DEV int bfrag_off(int tile, int ks, int col, int hi) {
  int p   = tile * 16 + col;
  int y   = p / WW;
  int x   = p - y * WW;
  int tap = ks * 2 + hi;
  int yy  = y + tap / 3 - 1;
  int xx  = x + tap % 3 - 1;
  bool valid = (tap < 9) && (p < HWP) &&
               ((unsigned)yy < (unsigned)HH) && ((unsigned)xx < (unsigned)WW);
  int sp = valid ? (yy * WW + xx) : HWP;  // slot 126 is zeroed
  return sp * 32;                         // 16 halves * 2 bytes
}

// ---------------------------------------------------------------------------
// Kernel 1: atom embedding  (N,1,7,18) -conv3x3-> softplus -> (N,16,7,18)
// ---------------------------------------------------------------------------
__global__ void __launch_bounds__(128)
k_embed(const float* __restrict__ nodes, const float* __restrict__ ew,
        const float* __restrict__ eb, float* __restrict__ atom,
        _Float16* __restrict__ atomT) {
  __shared__ float in[HH][WW];
  __shared__ float w[16][9];
  int n = blockIdx.x, t = threadIdx.x;
  for (int i = t; i < HWP; i += 128) in[i / WW][i % WW] = nodes[(size_t)n * HWP + i];
  for (int i = t; i < 144; i += 128) w[i / 9][i % 9] = ew[i];
  __syncthreads();
  for (int i = t; i < 16 * HWP; i += 128) {
    int c = i / HWP, p = i % HWP;
    int y = p / WW, x = p % WW;
    float acc = eb[c];
#pragma unroll
    for (int tap = 0; tap < 9; ++tap) {
      int yy = y + tap / 3 - 1, xx = x + tap % 3 - 1;
      if ((unsigned)yy < (unsigned)HH && (unsigned)xx < (unsigned)WW)
        acc += w[c][tap] * in[yy][xx];
    }
    float v = sp_f(acc);
    atom[(size_t)n * (16 * HWP) + i]            = v;
    atomT[(size_t)n * (PPAD * 16) + p * 16 + c] = (_Float16)v;
  }
  for (int i = t; i < 2 * 16; i += 128)  // zero pad slots 126,127
    atomT[(size_t)n * (PPAD * 16) + HWP * 16 + i] = (_Float16)0.f;
}

// ---------------------------------------------------------------------------
// Kernel 2: per-layer weight repack f32 -> f16 [tap][cout][cin] (+zero tap 9)
// ---------------------------------------------------------------------------
__global__ void k_weights_f16(const float* __restrict__ nw,
                              const float* __restrict__ ew,
                              const float* __restrict__ lw,
                              _Float16* __restrict__ nwT,
                              _Float16* __restrict__ ewT,
                              _Float16* __restrict__ lwT) {
  int i0 = blockIdx.x * blockDim.x + threadIdx.x;
  int stride = gridDim.x * blockDim.x;
  for (int idx = i0; idx < 10 * 16 * 16; idx += stride) {
    int tap = idx / 256, r = idx % 256, co = r / 16, ci = r % 16;
    float a = 0.f, b = 0.f;
    if (tap < 9) {
      int src = ((co * 16 + ci) * 3 + tap / 3) * 3 + tap % 3;
      a = nw[src];
      b = ew[src];
    }
    nwT[idx] = (_Float16)a;
    ewT[idx] = (_Float16)b;
  }
  for (int idx = i0; idx < 10 * 32 * 16; idx += stride) {
    int tap = idx / 512, r = idx % 512, co = r / 16, ci = r % 16;
    float a = 0.f;
    if (tap < 9) a = lw[((co * 16 + ci) * 3 + tap / 3) * 3 + tap % 3];
    lwT[idx] = (_Float16)a;
  }
}

// ---------------------------------------------------------------------------
// Kernel 3: node conv  th = conv2d(atom, nw)   -> thT f16 [node][pos][cout]
// One workgroup (4 waves) per node; 8 position tiles; 5 WMMA per tile.
// ---------------------------------------------------------------------------
__global__ void __launch_bounds__(128)
k_nodeconv(const _Float16* __restrict__ atomT, const _Float16* __restrict__ nwT,
           _Float16* __restrict__ thT) {
  __shared__ __align__(16) _Float16 fin[PPAD * 16];
  __shared__ __align__(16) _Float16 wl[10 * 16 * 16];
  int n = blockIdx.x, t = threadIdx.x;
  {
    const v8h* s = (const v8h*)(atomT + (size_t)n * (PPAD * 16));
    v8h* d = (v8h*)fin;
    for (int i = t; i < PPAD * 2; i += 128) d[i] = s[i];
  }
  for (int i = t; i < 10 * 16 * 16; i += 128) wl[i] = nwT[i];
  int wave = t >> 5, lane = t & 31;
  int col = lane & 15, hi = lane >> 4;
  int boff[2][5];
#pragma unroll
  for (int tt = 0; tt < 2; ++tt)
#pragma unroll
    for (int ks = 0; ks < 5; ++ks) boff[tt][ks] = bfrag_off(wave * 2 + tt, ks, col, hi);
  __syncthreads();
#pragma unroll
  for (int tt = 0; tt < 2; ++tt) {
    int tile = wave * 2 + tt;
    v8f acc = zero8();
#pragma unroll
    for (int ks = 0; ks < 5; ++ks)
      acc = wmma16(load_a_frag(wl, ks, 0, 16, lane), load_b_at(fin, boff[tt][ks]), acc);
    int p = tile * 16 + col;  // D: M = vgpr + 8*hi (cout), N = col (pos)
    v8h o;
#pragma unroll
    for (int v = 0; v < 8; ++v) o[v] = (_Float16)acc[v];
    *(v8h*)(thT + (size_t)n * (PPAD * 16) + p * 16 + hi * 8) = o;
  }
}

// ---------------------------------------------------------------------------
// Kernel 4: init scatter buffer (out = atom) and zero BN stats
// ---------------------------------------------------------------------------
__global__ void k_prepout(const float* __restrict__ atom, float* __restrict__ outb,
                          float* __restrict__ stats) {
  size_t i = (size_t)blockIdx.x * blockDim.x + threadIdx.x;
  size_t tot = (size_t)NNODE * 16 * HWP;
  size_t stride = (size_t)gridDim.x * blockDim.x;
  for (; i < tot; i += stride) outb[i] = atom[i];
  if (blockIdx.x == 0 && threadIdx.x < 32) stats[threadIdx.x] = 0.f;
}

// ---------------------------------------------------------------------------
// Kernel 5: fused edge pipeline (the hot kernel).
//   z   = elu( th[es] * conv(atom[et], ew) )
//   h   = conv(z, lw) + lb  (32 ch)
//   msg = sigmoid(h[:16]) * softplus(h[16:])
//   outb[es] += msg        (atomic scatter-add)
// One workgroup per edge; 30 WMMAs per wave (10 conv1 + 20 conv2).
// ---------------------------------------------------------------------------
__global__ void __launch_bounds__(128)
k_edge(const _Float16* __restrict__ atomT, const _Float16* __restrict__ thT,
       const _Float16* __restrict__ ewT, const _Float16* __restrict__ lwT,
       const int* __restrict__ es, const int* __restrict__ et,
       const float* __restrict__ lb, float* __restrict__ outb) {
  __shared__ __align__(16) _Float16 fin[PPAD * 16];
  __shared__ __align__(16) _Float16 fz[PPAD * 16];
  __shared__ __align__(16) _Float16 wew[10 * 16 * 16];
  __shared__ __align__(16) _Float16 wlw[10 * 32 * 16];
  int e = blockIdx.x, t = threadIdx.x;
  int srcN = es[e], tgtN = et[e];
  {
    const v8h* s = (const v8h*)(atomT + (size_t)tgtN * (PPAD * 16));
    v8h* d = (v8h*)fin;
    for (int i = t; i < PPAD * 2; i += 128) d[i] = s[i];
  }
  for (int i = t; i < 10 * 16 * 16; i += 128) wew[i] = ewT[i];
  for (int i = t; i < 10 * 32 * 16; i += 128) wlw[i] = lwT[i];

  int wave = t >> 5, lane = t & 31;
  int col = lane & 15, hi = lane >> 4;
  // Per-lane shifted-position offsets; identical for conv1 (fin) and conv2 (fz).
  int boff[2][5];
  int pPos[2];
#pragma unroll
  for (int tt = 0; tt < 2; ++tt) {
    pPos[tt] = (wave * 2 + tt) * 16 + col;
#pragma unroll
    for (int ks = 0; ks < 5; ++ks) boff[tt][ks] = bfrag_off(wave * 2 + tt, ks, col, hi);
  }
  // Gate values (produced by k_nodeconv): load early to hide latency.
  const _Float16* thBase = thT + (size_t)srcN * (PPAD * 16) + hi * 8;
  v8h thv[2];
#pragma unroll
  for (int tt = 0; tt < 2; ++tt)
    thv[tt] = *(const v8h*)(thBase + pPos[tt] * 16);
  __syncthreads();

  // ---- conv1 (16->16) + elu(th*z) -> fz ----
#pragma unroll
  for (int tt = 0; tt < 2; ++tt) {
    v8f acc = zero8();
#pragma unroll
    for (int ks = 0; ks < 5; ++ks)
      acc = wmma16(load_a_frag(wew, ks, 0, 16, lane), load_b_at(fin, boff[tt][ks]), acc);
    v8h zo;
#pragma unroll
    for (int v = 0; v < 8; ++v)
      zo[v] = (_Float16)elu_f((float)thv[tt][v] * acc[v]);
    *(v8h*)(fz + pPos[tt] * 16 + hi * 8) = zo;
  }
  __syncthreads();

  // ---- conv2 (16->32) + bias + gate + scatter ----
  const v4f* lb4 = (const v4f*)lb;
  v4f fb0 = lb4[hi * 2], fb1 = lb4[hi * 2 + 1];      // filter bias c=hi*8..+7
  v4f cb0 = lb4[4 + hi * 2], cb1 = lb4[4 + hi * 2 + 1];  // core bias
#pragma unroll
  for (int tt = 0; tt < 2; ++tt) {
    v8f accF = zero8();  // cout 0..15  (filter)
    v8f accC = zero8();  // cout 16..31 (core)
#pragma unroll
    for (int ks = 0; ks < 5; ++ks) {
      v16h b = load_b_at(fz, boff[tt][ks]);
      accF = wmma16(load_a_frag(wlw, ks, 0, 32, lane), b, accF);
      accC = wmma16(load_a_frag(wlw, ks, 16, 32, lane), b, accC);
    }
    int p = pPos[tt];
    if (p < HWP) {
      float* obase = outb + ((size_t)srcN * 16 + hi * 8) * HWP + p;
#pragma unroll
      for (int v = 0; v < 8; ++v) {
        float f = accF[v] + (v < 4 ? fb0[v & 3] : fb1[v & 3]);
        float g = accC[v] + (v < 4 ? cb0[v & 3] : cb1[v & 3]);
        float msg = sig_f(f) * sp_f(g);
        atomicAdd(obase + (size_t)v * HWP, msg);
      }
    }
  }
}

// ---------------------------------------------------------------------------
// Kernel 6: per-channel BN statistics (sum, sumsq) over (N, HW)
// blockIdx = chunk*16 + channel
// ---------------------------------------------------------------------------
__global__ void __launch_bounds__(128)
k_bnstats(const float* __restrict__ outb, float* __restrict__ stats) {
  int c = blockIdx.x & 15;
  int chunk = blockIdx.x >> 4;
  int nchunks = gridDim.x >> 4;
  float s = 0.f, s2 = 0.f;
  for (int n = chunk; n < NNODE; n += nchunks) {
    const float* base = outb + ((size_t)n * 16 + c) * HWP;
    for (int p = threadIdx.x; p < HWP; p += blockDim.x) {
      float v = base[p];
      s += v;
      s2 += v * v;
    }
  }
  __shared__ float rs[2][4];
  int lane = threadIdx.x & 31, wv = threadIdx.x >> 5;
  for (int off = 16; off; off >>= 1) {
    s += __shfl_down(s, off, 32);
    s2 += __shfl_down(s2, off, 32);
  }
  if (lane == 0) { rs[0][wv] = s; rs[1][wv] = s2; }
  __syncthreads();
  if (wv == 0 && lane == 0) {
    s = rs[0][0] + rs[0][1] + rs[0][2] + rs[0][3];
    s2 = rs[1][0] + rs[1][1] + rs[1][2] + rs[1][3];
    atomicAdd(&stats[c], s);
    atomicAdd(&stats[16 + c], s2);
  }
}

// ---------------------------------------------------------------------------
// Kernel 7: apply BN + residual; refresh atom f32 and atomT f16
// ---------------------------------------------------------------------------
__global__ void k_bnapply(const float* __restrict__ outb,
                          const float* __restrict__ stats,
                          const float* __restrict__ gg,
                          const float* __restrict__ bb,
                          float* __restrict__ atom, _Float16* __restrict__ atomT) {
  const float inv = 1.f / ((float)NNODE * (float)HWP);
  size_t tot = (size_t)NNODE * 16 * HWP;
  size_t stride = (size_t)gridDim.x * blockDim.x;
  for (size_t i = (size_t)blockIdx.x * blockDim.x + threadIdx.x; i < tot; i += stride) {
    int p = (int)(i % HWP);
    int c = (int)((i / HWP) % 16);
    size_t n = i / (16 * HWP);
    float m = stats[c] * inv;
    float var = stats[16 + c] * inv - m * m;
    float xn = gg[c] * (outb[i] - m) * rsqrtf(var + 1e-5f) + bb[c];
    float v = atom[i] + xn;
    atom[i] = v;
    atomT[n * (PPAD * 16) + (size_t)p * 16 + c] = (_Float16)v;
  }
}

// ---------------------------------------------------------------------------
// Tail kernels (tiny): segment mean, ctf conv, space branch, MLP head
// ---------------------------------------------------------------------------
__global__ void k_zero(float* p, int n) {
  for (int i = blockIdx.x * blockDim.x + threadIdx.x; i < n; i += gridDim.x * blockDim.x)
    p[i] = 0.f;
}

__global__ void k_segsum(const float* __restrict__ atom, const int* __restrict__ gidx,
                         float* __restrict__ crys) {
  size_t tot = (size_t)NNODE * 16 * HWP;
  size_t stride = (size_t)gridDim.x * blockDim.x;
  for (size_t i = (size_t)blockIdx.x * blockDim.x + threadIdx.x; i < tot; i += stride) {
    size_t n = i / (16 * HWP);
    atomicAdd(crys + (size_t)gidx[n] * (16 * HWP) + (i % (16 * HWP)), atom[i]);
  }
}

__global__ void __launch_bounds__(128)
k_ctf(const float* __restrict__ crys, const int* __restrict__ counts,
      const float* __restrict__ w, const float* __restrict__ bias,
      float* __restrict__ flat) {
  __shared__ float in[16][HH][WW];
  __shared__ float wl[16][16][9];
  int g = blockIdx.x, t = threadIdx.x;
  float invc = 1.f / (float)counts[g];
  for (int i = t; i < 16 * HWP; i += 128) {
    int c = i / HWP, p = i % HWP;
    in[c][p / WW][p % WW] = crys[(size_t)g * (16 * HWP) + i] * invc;
  }
  for (int i = t; i < 2304; i += 128) wl[i / 144][(i / 9) % 16][i % 9] = w[i];
  __syncthreads();
  for (int o = t; o < 16 * HWP; o += 128) {
    int co = o / HWP, p = o % HWP;
    int y = p / WW, x = p % WW;
    float acc = bias[co];
    for (int ci = 0; ci < 16; ++ci)
#pragma unroll
      for (int tap = 0; tap < 9; ++tap) {
        int yy = y + tap / 3 - 1, xx = x + tap % 3 - 1;
        if ((unsigned)yy < (unsigned)HH && (unsigned)xx < (unsigned)WW)
          acc += wl[co][ci][tap] * in[ci][yy][xx];
      }
    flat[(size_t)g * (16 * HWP) + o] = sp_f(acc);
  }
}

__global__ void __launch_bounds__(256)
k_space1(const int* __restrict__ nsites, const int* __restrict__ sg,
         const float* __restrict__ w1, const float* __restrict__ b1,
         const float* __restrict__ w2, const float* __restrict__ b2,
         const float* __restrict__ cw, const float* __restrict__ cb,
         float* __restrict__ sepre) {
  __shared__ float outer[16][16];
  __shared__ float row[16], colv[16];
  int g = blockIdx.x, t = threadIdx.x;
  float s0 = (float)nsites[g], s1 = (float)sg[g];
  if (t < 16) {
    row[t]  = s0 * w1[t * 2] + s1 * w1[t * 2 + 1] + b1[t];
    colv[t] = s0 * w2[t * 2] + s1 * w2[t * 2 + 1] + b2[t];
  }
  __syncthreads();
  if (t < 256) outer[t / 16][t % 16] = row[t / 16] * colv[t % 16];
  __syncthreads();
  for (int i = t; i < 16 * 256; i += 256) {
    int c = i / 256, p = i % 256, y = p / 16, x = p % 16;
    float acc = cb[c];
#pragma unroll
    for (int tap = 0; tap < 9; ++tap) {
      int yy = y + tap / 3 - 1, xx = x + tap % 3 - 1;
      if ((unsigned)yy < 16u && (unsigned)xx < 16u) acc += cw[c * 9 + tap] * outer[yy][xx];
    }
    sepre[((size_t)g * 16 + c) * 256 + p] = acc;
  }
}

__global__ void __launch_bounds__(256)
k_chstats(const float* __restrict__ x, float* __restrict__ stats, int S) {
  int c = blockIdx.x;
  float s = 0.f, s2 = 0.f;
  for (int i = threadIdx.x; i < NB * S; i += blockDim.x) {
    int b = i / S, p = i % S;
    float v = x[((size_t)b * 16 + c) * S + p];
    s += v;
    s2 += v * v;
  }
  __shared__ float rs[2][8];
  int lane = threadIdx.x & 31, wv = threadIdx.x >> 5;
  for (int off = 16; off; off >>= 1) {
    s += __shfl_down(s, off, 32);
    s2 += __shfl_down(s2, off, 32);
  }
  if (lane == 0) { rs[0][wv] = s; rs[1][wv] = s2; }
  __syncthreads();
  if (threadIdx.x == 0) {
    float a = 0.f, a2 = 0.f;
    for (int i = 0; i < 8; ++i) { a += rs[0][i]; a2 += rs[1][i]; }
    stats[c] = a;
    stats[16 + c] = a2;
  }
}

__global__ void k_bn_elem(const float* __restrict__ x, const float* __restrict__ stats,
                          const float* __restrict__ g, const float* __restrict__ b,
                          float* __restrict__ y, int S, int doSoftplus) {
  float inv = 1.f / ((float)NB * (float)S);
  int tot = NB * 16 * S;
  for (int i = blockIdx.x * blockDim.x + threadIdx.x; i < tot; i += gridDim.x * blockDim.x) {
    int c = (i / S) % 16;
    float m = stats[c] * inv;
    float var = stats[16 + c] * inv - m * m;
    float v = g[c] * (x[i] - m) * rsqrtf(var + 1e-5f) + b[c];
    y[i] = doSoftplus ? sp_f(v) : v;
  }
}

__global__ void __launch_bounds__(256)
k_space_conv2(const float* __restrict__ semb, const float* __restrict__ w,
              const float* __restrict__ bias, float* __restrict__ ypre) {
  __shared__ float in[16][16][16];
  __shared__ float wl[16][16][9];
  int g = blockIdx.x, t = threadIdx.x;
  for (int i = t; i < 4096; i += 256)
    in[i / 256][(i / 16) % 16][i % 16] = semb[(size_t)g * 4096 + i];
  for (int i = t; i < 2304; i += 256) wl[i / 144][(i / 9) % 16][i % 9] = w[i];
  __syncthreads();
  for (int o = t; o < 4096; o += 256) {
    int co = o / 256, p = o % 256, y = p / 16, x = p % 16;
    float acc = bias[co];
    for (int ci = 0; ci < 16; ++ci)
#pragma unroll
      for (int tap = 0; tap < 9; ++tap) {
        int yy = y + tap / 3 - 1, xx = x + tap % 3 - 1;
        if ((unsigned)yy < 16u && (unsigned)xx < 16u) acc += wl[co][ci][tap] * in[ci][yy][xx];
      }
    ypre[(size_t)g * 4096 + o] = acc;
  }
}

__global__ void k_pool(const float* __restrict__ ypre, const float* __restrict__ stats,
                       const float* __restrict__ g, const float* __restrict__ b,
                       float* __restrict__ sflat) {
  int gg = blockIdx.x, t = threadIdx.x;
  if (t < 144) {
    int c = t / 9, oy = (t % 9) / 3, ox = t % 3;
    float inv = 1.f / ((float)NB * 256.f);
    float m = stats[c] * inv;
    float var = stats[16 + c] * inv - m * m;
    float rs = rsqrtf(var + 1e-5f);
    float mx = -1e30f;
    for (int dy = 0; dy < 5; ++dy)
      for (int dx = 0; dx < 5; ++dx) {
        int y = oy * 5 + dy, x = ox * 5 + dx;
        float v = g[c] * (ypre[((size_t)gg * 16 + c) * 256 + y * 16 + x] - m) * rs + b[c];
        mx = fmaxf(mx, v);
      }
    sflat[gg * 144 + t] = sp_f(mx);
  }
}

__global__ void __launch_bounds__(128)
k_head(const float* __restrict__ flat, const float* __restrict__ sflat,
       const float* __restrict__ fc1w, const float* __restrict__ fc1b,
       const float* __restrict__ fc2w, const float* __restrict__ fc2b,
       const float* __restrict__ regw, const float* __restrict__ regb,
       const float* __restrict__ fc3w, const float* __restrict__ fc3b,
       const float* __restrict__ eps, float* __restrict__ out) {
  __shared__ float red[128];
  __shared__ float h1[32], h2[32];
  int g = blockIdx.x, t = threadIdx.x;
  int j = t >> 2, s = t & 3;
  const float* f = flat + (size_t)g * 2016;
  float acc = 0.f;
  for (int k = s; k < 2016; k += 4) acc += f[k] * fc1w[(size_t)j * 2016 + k];
  red[t] = acc;
  __syncthreads();
  if (s == 0) h1[j] = sp_f(red[t] + red[t + 1] + red[t + 2] + red[t + 3] + fc1b[j]);
  __syncthreads();
  if (t < 32) {
    float a = fc2b[t];
    for (int k = 0; k < 32; ++k) a += fc2w[t * 32 + k] * h1[k];
    h2[t] = sp_f(a);
  }
  __syncthreads();
  if (t == 0) {
    float o = regb[0];
    for (int k = 0; k < 32; ++k) o += regw[k] * h2[k];
    float o2 = fc3b[0];
    const float* sf = sflat + g * 144;
    for (int k = 0; k < 144; ++k) o2 += fc3w[k] * sf[k];
    out[g] = o + eps[0] * o2;
  }
}

// ---------------------------------------------------------------------------
// Host launch
// ---------------------------------------------------------------------------
extern "C" void kernel_launch(void* const* d_in, const int* in_sizes, int n_in,
                              void* d_out, int out_size, void* d_ws, size_t ws_size,
                              hipStream_t stream) {
  (void)in_sizes; (void)n_in; (void)out_size; (void)ws_size;
  const float* nodes   = (const float*)d_in[0];
  const int*   esrc    = (const int*)d_in[1];
  const int*   etgt    = (const int*)d_in[2];
  const int*   gidx    = (const int*)d_in[3];
  const int*   counts  = (const int*)d_in[4];
  const int*   nsites  = (const int*)d_in[5];
  const int*   sgroups = (const int*)d_in[6];
  const float* emb_w   = (const float*)d_in[7];
  const float* emb_b   = (const float*)d_in[8];
  const float* cl_nw   = (const float*)d_in[9];
  const float* cl_ew   = (const float*)d_in[10];
  const float* cl_lw   = (const float*)d_in[11];
  const float* cl_lb   = (const float*)d_in[12];
  const float* cl_bg   = (const float*)d_in[13];
  const float* cl_bb   = (const float*)d_in[14];
  const float* se_w1   = (const float*)d_in[15];
  const float* se_b1   = (const float*)d_in[16];
  const float* se_w2   = (const float*)d_in[17];
  const float* se_b2   = (const float*)d_in[18];
  const float* se_cw   = (const float*)d_in[19];
  const float* se_cb   = (const float*)d_in[20];
  const float* se_bg   = (const float*)d_in[21];
  const float* se_bb   = (const float*)d_in[22];
  const float* sf_cw   = (const float*)d_in[23];
  const float* sf_cb   = (const float*)d_in[24];
  const float* sf_bg   = (const float*)d_in[25];
  const float* sf_bb   = (const float*)d_in[26];
  const float* ctf_w   = (const float*)d_in[27];
  const float* ctf_b   = (const float*)d_in[28];
  const float* fc1w    = (const float*)d_in[29];
  const float* fc1b    = (const float*)d_in[30];
  const float* fc2w    = (const float*)d_in[31];
  const float* fc2b    = (const float*)d_in[32];
  const float* regw    = (const float*)d_in[33];
  const float* regb    = (const float*)d_in[34];
  const float* fc3w    = (const float*)d_in[35];
  const float* fc3b    = (const float*)d_in[36];
  const float* eps     = (const float*)d_in[37];
  float* out = (float*)d_out;

  char* base = (char*)d_ws;
  size_t off = 0;
  auto alloc = [&](size_t bytes) -> void* {
    void* p = base + off;
    off = (off + bytes + 255) & ~(size_t)255;
    return p;
  };
  float*    atom  = (float*)alloc((size_t)NNODE * 16 * HWP * 4);
  _Float16* atomT = (_Float16*)alloc((size_t)NNODE * PPAD * 16 * 2);
  _Float16* thT   = (_Float16*)alloc((size_t)NNODE * PPAD * 16 * 2);
  float*    outb  = (float*)alloc((size_t)NNODE * 16 * HWP * 4);
  float*    stats = (float*)alloc(32 * 4);
  _Float16* nwT   = (_Float16*)alloc(10 * 16 * 16 * 2);
  _Float16* ewT   = (_Float16*)alloc(10 * 16 * 16 * 2);
  _Float16* lwT   = (_Float16*)alloc(10 * 32 * 16 * 2);
  float*    crys  = (float*)alloc((size_t)NB * 16 * HWP * 4);
  float*    flat  = (float*)alloc((size_t)NB * 16 * HWP * 4);
  float*    sepre = (float*)alloc((size_t)NB * 16 * 256 * 4);
  float*    semb  = (float*)alloc((size_t)NB * 16 * 256 * 4);
  float*    ypre  = (float*)alloc((size_t)NB * 16 * 256 * 4);
  float*    sst1  = (float*)alloc(32 * 4);
  float*    sst2  = (float*)alloc(32 * 4);
  float*    sflat = (float*)alloc((size_t)NB * 144 * 4);

  k_embed<<<NNODE, 128, 0, stream>>>(nodes, emb_w, emb_b, atom, atomT);

  for (int l = 0; l < 3; ++l) {
    const float* nw = cl_nw + (size_t)l * 2304;
    const float* ew = cl_ew + (size_t)l * 2304;
    const float* lw = cl_lw + (size_t)l * 4608;
    const float* lb = cl_lb + (size_t)l * 32;
    const float* bg = cl_bg + (size_t)l * 16;
    const float* bb = cl_bb + (size_t)l * 16;
    k_weights_f16<<<8, 256, 0, stream>>>(nw, ew, lw, nwT, ewT, lwT);
    k_nodeconv<<<NNODE, 128, 0, stream>>>(atomT, nwT, thT);
    k_prepout<<<2048, 256, 0, stream>>>(atom, outb, stats);
    k_edge<<<NEDGE, 128, 0, stream>>>(atomT, thT, ewT, lwT, esrc, etgt, lb, outb);
    k_bnstats<<<16 * 64, 128, 0, stream>>>(outb, stats);
    k_bnapply<<<2048, 256, 0, stream>>>(outb, stats, bg, bb, atom, atomT);
  }

  k_zero<<<128, 256, 0, stream>>>(crys, NB * 16 * HWP);
  k_segsum<<<2048, 256, 0, stream>>>(atom, gidx, crys);
  k_ctf<<<NB, 128, 0, stream>>>(crys, counts, ctf_w, ctf_b, flat);

  k_space1<<<NB, 256, 0, stream>>>(nsites, sgroups, se_w1, se_b1, se_w2, se_b2,
                                   se_cw, se_cb, sepre);
  k_chstats<<<16, 256, 0, stream>>>(sepre, sst1, 256);
  k_bn_elem<<<256, 256, 0, stream>>>(sepre, sst1, se_bg, se_bb, semb, 256, 1);
  k_space_conv2<<<NB, 256, 0, stream>>>(semb, sf_cw, sf_cb, ypre);
  k_chstats<<<16, 256, 0, stream>>>(ypre, sst2, 256);
  k_pool<<<NB, 160, 0, stream>>>(ypre, sst2, sf_bg, sf_bb, sflat);

  k_head<<<NB, 128, 0, stream>>>(flat, sflat, fc1w, fc1b, fc2w, fc2b, regw, regb,
                                 fc3w, fc3b, eps, out);
}
